// SAINT_64141041598711
// MI455X (gfx1250) — compile-verified, matrix operand order checked by reading, not run on hardware
//
#include <hip/hip_runtime.h>

// ---------------------------------------------------------------------------
// Types for CDNA5 WMMA
// ---------------------------------------------------------------------------
typedef __attribute__((ext_vector_type(16))) __bf16 v16bf;
typedef __attribute__((ext_vector_type(8)))  __bf16 v8bf;
typedef __attribute__((ext_vector_type(8)))  float  v8f;

__device__ __forceinline__ __bf16 f2bf(float f) { return (__bf16)f; }

__device__ __forceinline__ v8f wmma_bf16(const v16bf& a, const v16bf& b, const v8f& c) {
  return __builtin_amdgcn_wmma_f32_16x16x32_bf16(false, a, false, b, (short)0, c, false, false);
}

__device__ __forceinline__ float gelu_exact(float x) {
  return 0.5f * x * (1.0f + erff(x * 0.7071067811865475f));
}

// 16-element bf16 fragment from LDS: row base p, elements {0..7, 16..23}
__device__ __forceinline__ v16bf ldsfrag(const __bf16* p) {
  v8bf lo = *(const v8bf*)(p);
  v8bf hi = *(const v8bf*)(p + 16);
  v16bf r;
#pragma unroll
  for (int i = 0; i < 8; ++i) { r[i] = lo[i]; r[8 + i] = hi[i]; }
  return r;
}

// A-fragment straight from global f32 row-major (p = &A[row][k0+ksel])
__device__ __forceinline__ v16bf gfragA(const float* p) {
  v16bf r;
#pragma unroll
  for (int i = 0; i < 8; ++i) r[i] = f2bf(p[i]);
#pragma unroll
  for (int i = 0; i < 8; ++i) r[8 + i] = f2bf(p[16 + i]);
  return r;
}

// ---------------------------------------------------------------------------
// CDNA5 async global->LDS copy (16B per lane), with compile-safe fallback
// ---------------------------------------------------------------------------
typedef __attribute__((__vector_size__(4 * sizeof(int)))) int i32x4;
typedef __attribute__((address_space(1))) i32x4 gi32x4;
typedef __attribute__((address_space(3))) i32x4 li32x4;

#if __has_builtin(__builtin_amdgcn_global_load_async_to_lds_b128)
#define HAVE_ASYNC_LDS 1
__device__ __forceinline__ void async_cp16(const __bf16* g, __bf16* l) {
  __builtin_amdgcn_global_load_async_to_lds_b128((gi32x4*)g, (li32x4*)l, 0, 0);
}
__device__ __forceinline__ void async_wait0() {
#if __has_builtin(__builtin_amdgcn_s_wait_asynccnt)
  __builtin_amdgcn_s_wait_asynccnt(0);
#else
  asm volatile("s_wait_asynccnt 0" ::: "memory");
#endif
}
#else
#define HAVE_ASYNC_LDS 0
__device__ __forceinline__ void async_cp16(const __bf16* g, __bf16* l) {
  *(v8bf*)l = *(const v8bf*)g;
}
__device__ __forceinline__ void async_wait0() {}
#endif

// ---------------------------------------------------------------------------
// Constants
// ---------------------------------------------------------------------------
#define Bsz   512
#define Nfeat 64
#define Dm    64
#define BND   (Bsz * Nfeat * Dm)     // 2,097,152
#define HEADSZ (512 * 512)           // 262,144 per head
#define WMAT  (4096 * 512)           // elems per IMHA weight matrix

// ---------------------------------------------------------------------------
// Embedding: h[b][n][d]
// ---------------------------------------------------------------------------
__global__ void embed_kernel(const float* __restrict__ x, const float* __restrict__ es,
                             const float* __restrict__ et, const float* __restrict__ nw,
                             const float* __restrict__ nb, float* __restrict__ h) {
  int idx = blockIdx.x * blockDim.x + threadIdx.x;
  if (idx >= BND) return;
  int d = idx & 63, n = (idx >> 6) & 63, b = idx >> 12;
  float v;
  if (n == 0)       v = es[(int)x[b * 64 + 0] * 64 + d];
  else if (n == 1)  v = et[(int)x[b * 64 + 1] * 64 + d];
  else              v = fmaxf(x[b * 64 + n] * nw[d] + nb[d], 0.0f);
  h[idx] = v;
}

// ---------------------------------------------------------------------------
// Repack MHA head weights (8,64,8) -> (64,64) head-concat, 6 matrices
// ---------------------------------------------------------------------------
struct SixPtr { const float* p[6]; };

__global__ void repack_mha(SixPtr sp, float* __restrict__ dst) {
  int idx = blockIdx.x * blockDim.x + threadIdx.x;
  if (idx >= 6 * 4096) return;
  int w = idx >> 12, rem = idx & 4095;
  int d = rem >> 6, j = rem & 63;
  int hh = j >> 3, k = j & 7;
  dst[idx] = sp.p[w][hh * 512 + d * 8 + k];
}

// ---------------------------------------------------------------------------
// f32 -> bf16 buffer conversion
// ---------------------------------------------------------------------------
__global__ void to_bf16(const float* __restrict__ src, __bf16* __restrict__ dst, int n) {
  int i = blockIdx.x * blockDim.x + threadIdx.x;
  if (i < n) dst[i] = f2bf(src[i]);
}

// ---------------------------------------------------------------------------
// IMHA weight transpose+convert: W[head](4096x512 f32) -> Wt[proj*8+head](512x4096 bf16)
// ---------------------------------------------------------------------------
__global__ void transpose_w(const float* __restrict__ wq, const float* __restrict__ wk,
                            const float* __restrict__ wv, __bf16* __restrict__ Wt) {
  __shared__ float tile[32][33];
  int z = blockIdx.z;                       // proj*8 + head
  int proj = z >> 3, head = z & 7;
  const float* W = (proj == 0 ? wq : (proj == 1 ? wk : wv)) + (size_t)head * WMAT;
  __bf16* dst = Wt + (size_t)z * WMAT;
  int k0 = blockIdx.x * 32, c0 = blockIdx.y * 32;
  int t = threadIdx.x, r8 = t >> 5, c = t & 31;
#pragma unroll
  for (int rr = 0; rr < 32; rr += 8)
    tile[rr + r8][c] = W[(size_t)(k0 + rr + r8) * 512 + c0 + c];
  __syncthreads();
#pragma unroll
  for (int rr = 0; rr < 32; rr += 8)
    dst[(size_t)(c0 + rr + r8) * 4096 + k0 + c] = f2bf(tile[c][rr + r8]);
}

// ---------------------------------------------------------------------------
// gemm64: C(32768x64) = A(32768x64 f32) @ W(64x64 f32), optional bias+gelu
// ---------------------------------------------------------------------------
__global__ void gemm64(const float* __restrict__ A, const float* __restrict__ W,
                       const float* __restrict__ bias, float* __restrict__ C, int mode) {
  __shared__ __align__(16) __bf16 Wt[64 * 64];   // transposed weight, bf16
  int tid = threadIdx.x;
#pragma unroll
  for (int i = 0; i < 16; ++i) {                 // stage W once
    int idx = i * 256 + tid;
    int r = idx >> 6, c = idx & 63;              // r = k, c = col
    Wt[c * 64 + r] = f2bf(W[r * 64 + c]);
  }
  __syncthreads();

  int wave = tid >> 5, lane = tid & 31;
  int row0 = blockIdx.x * 128 + wave * 16;
  int m = lane & 15, half = lane >> 4, ksel = half * 8;

  v8f acc[4] = {};
#pragma unroll
  for (int k0 = 0; k0 < 64; k0 += 32) {
    v16bf a = gfragA(A + (size_t)(row0 + m) * 64 + k0 + ksel);
#pragma unroll
    for (int ct = 0; ct < 4; ++ct) {
      v16bf bf = ldsfrag(&Wt[(ct * 16 + m) * 64 + k0 + ksel]);
      acc[ct] = wmma_bf16(a, bf, acc[ct]);
    }
  }
#pragma unroll
  for (int ct = 0; ct < 4; ++ct) {
    int col = ct * 16 + m;
    float bv = (mode == 1) ? bias[col] : 0.0f;
#pragma unroll
    for (int i = 0; i < 8; ++i) {
      float v = acc[ct][i] + bv;
      if (mode == 1) v = gelu_exact(v);
      C[(size_t)(row0 + half * 8 + i) * 64 + col] = v;
    }
  }
}

// ---------------------------------------------------------------------------
// MHA attention per (batch, head). 64 threads, everything in LDS.
// ---------------------------------------------------------------------------
__global__ void mha_attn(const float* __restrict__ Qc, const float* __restrict__ Kc,
                         const float* __restrict__ Vc, float* __restrict__ cat) {
  __shared__ float qs[64 * 8], ks[64 * 8], vs[64 * 8];
  int b = blockIdx.x >> 3, hd = blockIdx.x & 7;
  int n = threadIdx.x;
  size_t base = (size_t)(b * 64 + n) * 64 + hd * 8;
#pragma unroll
  for (int k = 0; k < 8; ++k) {
    qs[n * 8 + k] = Qc[base + k];
    ks[n * 8 + k] = Kc[base + k];
    vs[n * 8 + k] = Vc[base + k];
  }
  __syncthreads();
  float mx = -1e30f;
  for (int mm = 0; mm < 64; ++mm) {
    float d = 0.f;
#pragma unroll
    for (int k = 0; k < 8; ++k) d += qs[n * 8 + k] * ks[mm * 8 + k];
    mx = fmaxf(mx, d * 0.125f);
  }
  float sum = 0.f, av[8] = {0.f,0.f,0.f,0.f,0.f,0.f,0.f,0.f};
  for (int mm = 0; mm < 64; ++mm) {
    float d = 0.f;
#pragma unroll
    for (int k = 0; k < 8; ++k) d += qs[n * 8 + k] * ks[mm * 8 + k];
    float e = expf(d * 0.125f - mx);
    sum += e;
#pragma unroll
    for (int k = 0; k < 8; ++k) av[k] += e * vs[mm * 8 + k];
  }
  float inv = 1.0f / sum;
#pragma unroll
  for (int k = 0; k < 8; ++k) cat[base + k] = av[k] * inv;
}

// ---------------------------------------------------------------------------
// LayerNorm(y)*g + b + h  -> h (in place). One wave per row of 64.
// ---------------------------------------------------------------------------
__global__ void ln_residual(const float* __restrict__ Y, const float* __restrict__ g,
                            const float* __restrict__ bb, float* __restrict__ H) {
  int wave = threadIdx.x >> 5, lane = threadIdx.x & 31;
  size_t row = (size_t)blockIdx.x * 8 + wave;
  const float* y = Y + row * 64;
  float* h = H + row * 64;
  float y0 = y[lane], y1 = y[lane + 32];
  float s = y0 + y1;
#pragma unroll
  for (int msk = 16; msk; msk >>= 1) s += __shfl_xor(s, msk, 32);
  float mean = s * (1.0f / 64.0f);
  float r0 = y0 - mean, r1 = y1 - mean;
  float v = r0 * r0 + r1 * r1;
#pragma unroll
  for (int msk = 16; msk; msk >>= 1) v += __shfl_xor(v, msk, 32);
  float inv = rsqrtf(v * (1.0f / 64.0f) + 1e-3f);
  h[lane]      += r0 * inv * g[lane]      + bb[lane];
  h[lane + 32] += r1 * inv * g[lane + 32] + bb[lane + 32];
}

// ---------------------------------------------------------------------------
// IMHA projections, pure-bf16 double-buffered async pipeline.
// A = h (512x4096 bf16), W = Wt (512x4096 bf16, pre-transposed), per (head,proj).
// Outputs bf16: Q, K normal; V transposed.
// ---------------------------------------------------------------------------
__global__ void imha_proj(const __bf16* __restrict__ Ab, const __bf16* __restrict__ Wtb,
                          __bf16* __restrict__ Q, __bf16* __restrict__ Kk,
                          __bf16* __restrict__ Vt) {
  __shared__ __align__(16) __bf16 As[2][128 * 32];
  __shared__ __align__(16) __bf16 Bs[2][128 * 32];
  int tid = threadIdx.x;
  int tx = blockIdx.x & 3, ty = blockIdx.x >> 2;
  int row0 = ty * 128, col0 = tx * 128;
  int head = blockIdx.y, proj = blockIdx.z;
  const __bf16* Wb = Wtb + (size_t)(proj * 8 + head) * WMAT;
  __bf16* C = (proj == 0 ? Q : (proj == 1 ? Kk : Vt)) + (size_t)head * HEADSZ;

  int wave = tid >> 5, lane = tid & 31;
  int rt0 = (wave & 3) * 2, ct0 = (wave >> 2) * 4;
  int m = lane & 15, half = lane >> 4, ksel = half * 8;

  // two 16B chunks per thread per tile (128x32 bf16 = 8KB)
  int r0c = tid >> 2,         co0 = (tid & 3) * 8;
  int r1c = (256 + tid) >> 2, co1 = ((256 + tid) & 3) * 8;

  auto issue = [&](int buf, int kk) {
    async_cp16(Ab + (size_t)(row0 + r0c) * 4096 + kk + co0, &As[buf][r0c * 32 + co0]);
    async_cp16(Ab + (size_t)(row0 + r1c) * 4096 + kk + co1, &As[buf][r1c * 32 + co1]);
    async_cp16(Wb + (size_t)(col0 + r0c) * 4096 + kk + co0, &Bs[buf][r0c * 32 + co0]);
    async_cp16(Wb + (size_t)(col0 + r1c) * 4096 + kk + co1, &Bs[buf][r1c * 32 + co1]);
  };

  v8f acc[2][4] = {};
  issue(0, 0);
  for (int kk = 0; kk < 4096; kk += 32) {
    int cur = (kk >> 5) & 1;
    async_wait0();
    __syncthreads();
    if (kk + 32 < 4096) issue(cur ^ 1, kk + 32);
    v16bf af[2], bf[4];
#pragma unroll
    for (int j = 0; j < 2; ++j) af[j] = ldsfrag(&As[cur][((rt0 + j) * 16 + m) * 32 + ksel]);
#pragma unroll
    for (int j = 0; j < 4; ++j) bf[j] = ldsfrag(&Bs[cur][((ct0 + j) * 16 + m) * 32 + ksel]);
#pragma unroll
    for (int j = 0; j < 2; ++j)
#pragma unroll
      for (int jj = 0; jj < 4; ++jj)
        acc[j][jj] = wmma_bf16(af[j], bf[jj], acc[j][jj]);
  }
#pragma unroll
  for (int j = 0; j < 2; ++j)
#pragma unroll
    for (int jj = 0; jj < 4; ++jj)
#pragma unroll
      for (int i = 0; i < 8; ++i) {
        int mg = row0 + (rt0 + j) * 16 + half * 8 + i;
        int ng = col0 + (ct0 + jj) * 16 + m;
        if (proj < 2) C[(size_t)mg * 512 + ng] = f2bf(acc[j][jj][i]);
        else          C[(size_t)ng * 512 + mg] = f2bf(acc[j][jj][i]);   // V transposed
      }
}

// ---------------------------------------------------------------------------
// NT GEMM per head: C f32 (512x512) = A bf16 (512x512) @ B bf16 (512x512)^T
// (contraction over last dim of both). Double-buffered async staging.
// ---------------------------------------------------------------------------
__global__ void gemm_nt512(const __bf16* __restrict__ Ah, const __bf16* __restrict__ Bh,
                           float* __restrict__ Ch) {
  __shared__ __align__(16) __bf16 As[2][128 * 32];
  __shared__ __align__(16) __bf16 Bs[2][128 * 32];
  int head = blockIdx.y;
  const __bf16* A = Ah + (size_t)head * HEADSZ;
  const __bf16* B = Bh + (size_t)head * HEADSZ;
  float* C = Ch + (size_t)head * HEADSZ;
  int tid = threadIdx.x;
  int tx = blockIdx.x & 3, ty = blockIdx.x >> 2;
  int row0 = ty * 128, col0 = tx * 128;
  int wave = tid >> 5, lane = tid & 31;
  int rt0 = (wave & 3) * 2, ct0 = (wave >> 2) * 4;
  int m = lane & 15, half = lane >> 4, ksel = half * 8;

  int r0c = tid >> 2,         co0 = (tid & 3) * 8;
  int r1c = (256 + tid) >> 2, co1 = ((256 + tid) & 3) * 8;

  auto issue = [&](int buf, int kk) {
    async_cp16(A + (size_t)(row0 + r0c) * 512 + kk + co0, &As[buf][r0c * 32 + co0]);
    async_cp16(A + (size_t)(row0 + r1c) * 512 + kk + co1, &As[buf][r1c * 32 + co1]);
    async_cp16(B + (size_t)(col0 + r0c) * 512 + kk + co0, &Bs[buf][r0c * 32 + co0]);
    async_cp16(B + (size_t)(col0 + r1c) * 512 + kk + co1, &Bs[buf][r1c * 32 + co1]);
  };

  v8f acc[2][4] = {};
  issue(0, 0);
  for (int kk = 0; kk < 512; kk += 32) {
    int cur = (kk >> 5) & 1;
    async_wait0();
    __syncthreads();
    if (kk + 32 < 512) issue(cur ^ 1, kk + 32);
    v16bf af[2], bf[4];
#pragma unroll
    for (int j = 0; j < 2; ++j) af[j] = ldsfrag(&As[cur][((rt0 + j) * 16 + m) * 32 + ksel]);
#pragma unroll
    for (int j = 0; j < 4; ++j) bf[j] = ldsfrag(&Bs[cur][((ct0 + j) * 16 + m) * 32 + ksel]);
#pragma unroll
    for (int j = 0; j < 2; ++j)
#pragma unroll
      for (int jj = 0; jj < 4; ++jj)
        acc[j][jj] = wmma_bf16(af[j], bf[jj], acc[j][jj]);
  }
#pragma unroll
  for (int j = 0; j < 2; ++j)
#pragma unroll
    for (int jj = 0; jj < 4; ++jj)
#pragma unroll
      for (int i = 0; i < 8; ++i)
        C[(size_t)(row0 + (rt0 + j) * 16 + half * 8 + i) * 512 +
          col0 + (ct0 + jj) * 16 + m] = acc[j][jj][i];
}

// ---------------------------------------------------------------------------
// Row softmax over 512 with scale 1/8; f32 in, bf16 out.
// ---------------------------------------------------------------------------
__global__ void softmax512(const float* __restrict__ S, __bf16* __restrict__ P) {
  __shared__ float red[8];
  int head = blockIdx.x >> 9, row = blockIdx.x & 511;
  size_t base = (size_t)head * HEADSZ + (size_t)row * 512;
  int t = threadIdx.x, wv = t >> 5, ln = t & 31;
  float a = S[base + t] * 0.125f, b = S[base + t + 256] * 0.125f;
  float mx = fmaxf(a, b);
#pragma unroll
  for (int msk = 16; msk; msk >>= 1) mx = fmaxf(mx, __shfl_xor(mx, msk, 32));
  if (!ln) red[wv] = mx;
  __syncthreads();
  mx = red[0];
#pragma unroll
  for (int i = 1; i < 8; ++i) mx = fmaxf(mx, red[i]);
  __syncthreads();
  float e0 = expf(a - mx), e1 = expf(b - mx);
  float s = e0 + e1;
#pragma unroll
  for (int msk = 16; msk; msk >>= 1) s += __shfl_xor(s, msk, 32);
  if (!ln) red[wv] = s;
  __syncthreads();
  s = 0.f;
#pragma unroll
  for (int i = 0; i < 8; ++i) s += red[i];
  float inv = 1.0f / s;
  P[base + t]       = f2bf(e0 * inv);
  P[base + t + 256] = f2bf(e1 * inv);
}

// ---------------------------------------------------------------------------
// IMHA concat: cat[b][n][h*8+k] = AV[h][b][n*8+k]
// ---------------------------------------------------------------------------
__global__ void imha_cat(const float* __restrict__ AV, float* __restrict__ cat) {
  int idx = blockIdx.x * blockDim.x + threadIdx.x;
  if (idx >= BND) return;
  int j = idx & 63, n = (idx >> 6) & 63, b = idx >> 12;
  int hh = j >> 3, k = j & 7;
  cat[idx] = AV[(size_t)hh * HEADSZ + (size_t)b * 512 + n * 8 + k];
}

__global__ void gather_out(const float* __restrict__ h, float* __restrict__ out) {
  int i = blockIdx.x * blockDim.x + threadIdx.x;
  if (i >= Bsz * Dm) return;
  out[i] = h[(size_t)(i >> 6) * 4096 + (i & 63)];
}

// ---------------------------------------------------------------------------
// Host launcher
// ---------------------------------------------------------------------------
extern "C" void kernel_launch(void* const* d_in, const int* in_sizes, int n_in,
                              void* d_out, int out_size, void* d_ws, size_t ws_size,
                              hipStream_t stream) {
  (void)in_sizes; (void)n_in; (void)out_size; (void)ws_size;
  int p = 0;
  const float* x         = (const float*)d_in[p++];
  const float* emb_stock = (const float*)d_in[p++];
  const float* emb_time  = (const float*)d_in[p++];
  const float* num_w     = (const float*)d_in[p++];
  const float* num_b     = (const float*)d_in[p++];
  const float *mha_wq[2], *mha_wk[2], *mha_wv[2], *mha_wo[2];
  for (int l = 0; l < 2; ++l) {
    mha_wq[l] = (const float*)d_in[p++];
    mha_wk[l] = (const float*)d_in[p++];
    mha_wv[l] = (const float*)d_in[p++];
    mha_wo[l] = (const float*)d_in[p++];
  }
  const float *imha_wq[2], *imha_wk[2], *imha_wv[2], *imha_wo[2];
  for (int l = 0; l < 2; ++l) {
    imha_wq[l] = (const float*)d_in[p++];
    imha_wk[l] = (const float*)d_in[p++];
    imha_wv[l] = (const float*)d_in[p++];
    imha_wo[l] = (const float*)d_in[p++];
  }
  const float *ffn_w[4], *ffn_b[4];
  for (int i = 0; i < 4; ++i) {
    ffn_w[i] = (const float*)d_in[p++];
    ffn_b[i] = (const float*)d_in[p++];
  }
  const float *ln_g[8], *ln_b[8];
  for (int i = 0; i < 8; ++i) {
    ln_g[i] = (const float*)d_in[p++];
    ln_b[i] = (const float*)d_in[p++];
  }

  // ---- workspace carve-up ----
  char* wsb = (char*)d_ws;
  size_t off = 0;
  auto carveF = [&](size_t n) { float* r = (float*)(wsb + off); off += n * 4; return r; };
  auto carveB = [&](size_t n) { __bf16* r = (__bf16*)(wsb + off); off += n * 2; return r; };
  float*  h    = carveF(BND);
  float*  t0   = carveF(BND);
  float*  t1   = carveF(BND);
  float*  t2   = carveF(BND);
  float*  t3   = carveF(BND);
  float*  Sb   = carveF(BND);
  float*  AVb  = carveF(BND);
  float*  Wrep = carveF(6 * 4096);
  __bf16* hb   = carveB(BND);
  __bf16* Qb   = carveB(BND);
  __bf16* Kb   = carveB(BND);
  __bf16* Vtb  = carveB(BND);
  __bf16* Pb   = carveB(BND);
  __bf16* Wt   = carveB((size_t)24 * WMAT);

  const int TB = 256;
  embed_kernel<<<BND / TB, TB, 0, stream>>>(x, emb_stock, emb_time, num_w, num_b, h);

  SixPtr sp;
  sp.p[0] = mha_wq[0]; sp.p[1] = mha_wk[0]; sp.p[2] = mha_wv[0];
  sp.p[3] = mha_wq[1]; sp.p[4] = mha_wk[1]; sp.p[5] = mha_wv[1];
  repack_mha<<<(6 * 4096 + TB - 1) / TB, TB, 0, stream>>>(sp, Wrep);

  for (int l = 0; l < 2; ++l) {
    const float* Wq = Wrep + (size_t)(l * 3 + 0) * 4096;
    const float* Wk = Wrep + (size_t)(l * 3 + 1) * 4096;
    const float* Wv = Wrep + (size_t)(l * 3 + 2) * 4096;

    // --- MHA ---
    gemm64<<<256, TB, 0, stream>>>(h, Wq, nullptr, t0, 0);
    gemm64<<<256, TB, 0, stream>>>(h, Wk, nullptr, t1, 0);
    gemm64<<<256, TB, 0, stream>>>(h, Wv, nullptr, t2, 0);
    mha_attn<<<4096, 64, 0, stream>>>(t0, t1, t2, t3);
    gemm64<<<256, TB, 0, stream>>>(t3, mha_wo[l], nullptr, t0, 0);
    ln_residual<<<4096, TB, 0, stream>>>(t0, ln_g[l], ln_b[l], h);

    // --- FFN A ---
    gemm64<<<256, TB, 0, stream>>>(h, ffn_w[l], ffn_b[l], t0, 1);
    ln_residual<<<4096, TB, 0, stream>>>(t0, ln_g[l + 1], ln_b[l + 1], h);

    // --- IMHA ---
    transpose_w<<<dim3(128, 16, 24), TB, 0, stream>>>(imha_wq[l], imha_wk[l], imha_wv[l], Wt);
    to_bf16<<<BND / TB, TB, 0, stream>>>(h, hb, BND);
    imha_proj<<<dim3(16, 8, 3), TB, 0, stream>>>(hb, Wt, Qb, Kb, Vtb);
    gemm_nt512<<<dim3(16, 8), TB, 0, stream>>>(Qb, Kb, Sb);
    softmax512<<<4096, TB, 0, stream>>>(Sb, Pb);
    gemm_nt512<<<dim3(16, 8), TB, 0, stream>>>(Pb, Vtb, AVb);
    imha_cat<<<BND / TB, TB, 0, stream>>>(AVb, t0);
    gemm64<<<256, TB, 0, stream>>>(t0, imha_wo[l], nullptr, t1, 0);
    ln_residual<<<4096, TB, 0, stream>>>(t1, ln_g[l + 2], ln_b[l + 2], h);

    // --- FFN B ---
    gemm64<<<256, TB, 0, stream>>>(h, ffn_w[l + 1], ffn_b[l + 1], t0, 1);
    ln_residual<<<4096, TB, 0, stream>>>(t0, ln_g[l + 3], ln_b[l + 3], h);
  }

  gather_out<<<(Bsz * Dm + TB - 1) / TB, TB, 0, stream>>>(h, (float*)d_out);
}
